// Fp6LlmLinear_48756468744613
// MI455X (gfx1250) — matrix operand order repaired and checked by reading, not compiled, not run on hardware
//
#include <hip/hip_runtime.h>

// FP6-e3m2 (stored as fp32) x fp16 fused dequant-GEMM for gfx1250 (MI455X).
// out[m,n] = fp16( scale[n] * sum_k x[m,k]*w6[n,k] ) + bias[n]   (fp16 add)
//
// Tiling: 64(M) x 128(N) x 32(K) per block, 256 threads = 8 wave32 waves
// arranged 2(M) x 4(N); each wave computes a 32x32 tile via 4x
// v_wmma_f32_16x16x32_f16 per k-step. Double-buffered LDS staging:
//   - x tile: GLOBAL_LOAD_ASYNC_TO_LDS_B128 (ASYNCcnt) when available
//   - w tile: register-staged fp32->fp16 conversion (co-executes with WMMA)

typedef __attribute__((ext_vector_type(16))) _Float16 v16h;
typedef __attribute__((ext_vector_type(8)))  _Float16 v8h;
typedef __attribute__((ext_vector_type(8)))  float    v8f;
typedef __attribute__((ext_vector_type(4)))  float    v4f;
typedef __attribute__((ext_vector_type(4)))  int      v4i;

#define BM   64
#define BN   128
#define BK   32
#define LDSK 40          // padded LDS row stride in halfs (80B: keeps 16B align)
#define PFK  6           // prefetch distance in k-tiles

#if __has_builtin(__builtin_amdgcn_global_load_async_to_lds_b128)
#define USE_ASYNC_LDS 1
#else
#define USE_ASYNC_LDS 0
#endif

// global (AS1) and LDS (AS3) pointer types for the async builtin
typedef __attribute__((address_space(1))) v4i* gv4i_t;
typedef __attribute__((address_space(3))) v4i* lv4i_t;

__device__ __forceinline__ void wait_asynccnt0() {
#if __has_builtin(__builtin_amdgcn_s_wait_asynccnt)
    __builtin_amdgcn_s_wait_asynccnt(0);
#else
    asm volatile("s_wait_asynccnt 0x0" ::: "memory");
#endif
}

__device__ __forceinline__ void async_copy16(const void* g, void* lds) {
#if USE_ASYNC_LDS
    // global AS shares the flat identity mapping -> inttoptr is safe;
    // LDS needs a true addrspacecast (generic -> AS3).
    __builtin_amdgcn_global_load_async_to_lds_b128(
        (gv4i_t)(uintptr_t)g, (lv4i_t)lds, 0, 0);
#endif
}

union AB16 { v16h v; v8h h[2]; };

__global__ __launch_bounds__(256) void fp6_linear_wmma(
    const _Float16* __restrict__ x,      // [M,K] fp16 row-major
    const float*    __restrict__ w,      // [N,K] fp32 (values on e3m2 grid)
    const _Float16* __restrict__ scales, // [N] fp16 dequant scale
    const _Float16* __restrict__ bias,   // [N] fp16
    _Float16*       __restrict__ out,    // [M,N] fp16
    int M, int N, int K)
{
    __shared__ _Float16 xs[2][BM * LDSK];
    __shared__ _Float16 ws[2][BN * LDSK];

    const int t    = threadIdx.x;        // 0..255
    const int lane = t & 31;
    const int wid  = t >> 5;             // 0..7
    const int wm   = wid & 1;            // wave M group: 2 x 32 rows
    const int wn   = wid >> 1;           // wave N group: 4 x 32 cols
    const int l16  = lane & 15;
    const int hi   = lane >> 4;          // half-wave select

    const int m0 = blockIdx.y * BM;
    const int n0 = blockIdx.x * BN;

    // --- global->LDS staging map ---
    // X tile: 64 rows x 32 halfs = 256 chunks of 8 halfs (16B)
    const int xrow = t >> 2;
    const int xchk = (t & 3) * 8;
    // W tile: 128 rows x 32 floats = 256 chunks of 16 floats (4x float4)
    const int wrow = t >> 1;
    const int wchk = (t & 1) * 16;

    const _Float16* xg = x + (size_t)(m0 + xrow) * K + xchk;
    const float*    wg = w + (size_t)(n0 + wrow) * K + wchk;

    // hoist epilogue operand loads: latency hidden behind the k-loop
    const int nc0 = n0 + wn * 32 + l16;
    const int nc1 = nc0 + 16;
    const float    sc0 = (float)scales[nc0];
    const float    sc1 = (float)scales[nc1];
    const _Float16 bv0 = bias[nc0];
    const _Float16 bv1 = bias[nc1];

    v8f acc[2][2] = {};

    const int NK = K / BK;

    // ---- prologue: stage k-tile 0 into buffer 0 ----
#if USE_ASYNC_LDS
    async_copy16(xg, &xs[0][xrow * LDSK + xchk]);
#else
    v8h xr = *(const v8h*)xg;
#endif
    v4f wr0 = *(const v4f*)(wg + 0);
    v4f wr1 = *(const v4f*)(wg + 4);
    v4f wr2 = *(const v4f*)(wg + 8);
    v4f wr3 = *(const v4f*)(wg + 12);

    {
#if !USE_ASYNC_LDS
        *(v8h*)&xs[0][xrow * LDSK + xchk] = xr;
#endif
        v8h h0, h1;
#pragma unroll
        for (int i = 0; i < 4; ++i) {
            h0[i]     = (_Float16)wr0[i];
            h0[i + 4] = (_Float16)wr1[i];
            h1[i]     = (_Float16)wr2[i];
            h1[i + 4] = (_Float16)wr3[i];
        }
        *(v8h*)&ws[0][wrow * LDSK + wchk]     = h0;
        *(v8h*)&ws[0][wrow * LDSK + wchk + 8] = h1;
    }
#if USE_ASYNC_LDS
    wait_asynccnt0();
#endif
    __syncthreads();

    int buf = 0;
    for (int kt = 0; kt < NK; ++kt) {
        const bool has_next = (kt + 1) < NK;

        // issue next tile's transfers early (hide HBM latency behind WMMA)
        if (has_next) {
#if USE_ASYNC_LDS
            async_copy16(xg + BK, &xs[buf ^ 1][xrow * LDSK + xchk]);
#else
            xr  = *(const v8h*)(xg + BK);
#endif
            wr0 = *(const v4f*)(wg + BK + 0);
            wr1 = *(const v4f*)(wg + BK + 4);
            wr2 = *(const v4f*)(wg + BK + 8);
            wr3 = *(const v4f*)(wg + BK + 12);
        }
        // L2 prefetch of the weight stream further ahead (global_prefetch_b8)
        if (kt + PFK < NK) {
            __builtin_prefetch((const void*)(wg + (size_t)PFK * BK), 0, 1);
        }

        // ---- fragment loads from LDS ----
        // A (16x32 f16): lanes 0-15 hold K 0..7 / 16..23; lanes 16-31 shifted
        AB16 A[2];
#pragma unroll
        for (int mt = 0; mt < 2; ++mt) {
            const _Float16* xp = &xs[buf][(wm * 32 + mt * 16 + l16) * LDSK];
            A[mt].h[0] = *(const v8h*)(xp + hi * 8);
            A[mt].h[1] = *(const v8h*)(xp + 16 + hi * 8);
        }
        // B (32x16 f16): lane l16 = output column n; K 0..7 / 16..23, lanes 16-31 shifted (same as A)
        AB16 B[2];
#pragma unroll
        for (int nt = 0; nt < 2; ++nt) {
            const _Float16* wp = &ws[buf][(wn * 32 + nt * 16 + l16) * LDSK];
            B[nt].h[0] = *(const v8h*)(wp + hi * 8);
            B[nt].h[1] = *(const v8h*)(wp + 16 + hi * 8);
        }

        // ---- 4x WMMA 16x16x32 f16 -> f32 acc ----
#pragma unroll
        for (int mt = 0; mt < 2; ++mt)
#pragma unroll
            for (int nt = 0; nt < 2; ++nt)
                acc[mt][nt] = __builtin_amdgcn_wmma_f32_16x16x32_f16(
                    false, A[mt].v, false, B[nt].v,
                    (short)0, acc[mt][nt], false, false);

        if (has_next) {
            // stage next W tile into the other buffer
            const int nb = buf ^ 1;
#if !USE_ASYNC_LDS
            *(v8h*)&xs[nb][xrow * LDSK + xchk] = xr;
#endif
            v8h h0, h1;
#pragma unroll
            for (int i = 0; i < 4; ++i) {
                h0[i]     = (_Float16)wr0[i];
                h0[i + 4] = (_Float16)wr1[i];
                h1[i]     = (_Float16)wr2[i];
                h1[i + 4] = (_Float16)wr3[i];
            }
            *(v8h*)&ws[nb][wrow * LDSK + wchk]     = h0;
            *(v8h*)&ws[nb][wrow * LDSK + wchk + 8] = h1;
#if USE_ASYNC_LDS
            wait_asynccnt0();      // x tile landed in LDS
#endif
            __syncthreads();
            buf = nb;
            xg += BK;
            wg += BK;
        }
    }

    // ---- epilogue: scale (f32), cast to f16, f16 bias add, store ----
    // C/D layout: VGPR v -> row (v + 8*hi), lane l16 -> col
#pragma unroll
    for (int nt = 0; nt < 2; ++nt) {
        const int   n  = (nt == 0) ? nc0 : nc1;
        const float s  = (nt == 0) ? sc0 : sc1;
        const _Float16 bv = (nt == 0) ? bv0 : bv1;
#pragma unroll
        for (int mt = 0; mt < 2; ++mt) {
            const int mb = m0 + wm * 32 + mt * 16 + hi * 8;
#pragma unroll
            for (int v = 0; v < 8; ++v) {
                _Float16 hv = (_Float16)(acc[mt][nt][v] * s);
                out[(size_t)(mb + v) * N + n] = (_Float16)(hv + bv);
            }
        }
    }
}

extern "C" void kernel_launch(void* const* d_in, const int* in_sizes, int n_in,
                              void* d_out, int out_size, void* d_ws, size_t ws_size,
                              hipStream_t stream) {
    const _Float16* x      = (const _Float16*)d_in[0];
    const float*    w      = (const float*)d_in[1];
    const _Float16* scales = (const _Float16*)d_in[2];
    const _Float16* bias   = (const _Float16*)d_in[3];
    _Float16*       out    = (_Float16*)d_out;

    const int N = in_sizes[2];               // 4096 (scales)
    const int K = in_sizes[1] / N;           // 4096
    const int M = in_sizes[0] / K;           // 256

    dim3 grid(N / BN, M / BM);               // 32 x 4 = 128 workgroups
    fp6_linear_wmma<<<grid, 256, 0, stream>>>(x, w, scales, bias, out, M, N, K);
}
